// LSTMCell_24953759990253
// MI455X (gfx1250) — matrix operand order repaired and checked by reading, not compiled, not run on hardware
//
#include <hip/hip_runtime.h>
#include <hip/hip_bf16.h>
#include <stdint.h>

// ---------------- problem constants ----------------
#define BATCH 16384
#define INF   512
#define HID   512
#define KTOT  1024            // IN + HID

// tile config
#define BM 128
#define BN 64
#define BK 32
#define NTHREADS 256          // 8 wave32 waves

typedef __attribute__((ext_vector_type(16))) __bf16 v16bf;
typedef __attribute__((ext_vector_type(8)))  __bf16 v8bf;
typedef __attribute__((ext_vector_type(2)))  __bf16 v2bf;
typedef __attribute__((ext_vector_type(8)))  float  v8f;

union AFrag { v16bf v; v8bf h[2]; };

#if __has_builtin(__builtin_amdgcn_global_load_async_to_lds_b128)
#define HAVE_ASYNC_LDS 1
#else
#define HAVE_ASYNC_LDS 0
#endif

// exact parameter types per hipcc diagnostic: pointer-to-int4 in AS1 / AS3
typedef __attribute__((__vector_size__(4 * sizeof(int)))) int vsi4;
typedef __attribute__((address_space(1))) vsi4 as1_vsi4;
typedef __attribute__((address_space(3))) vsi4 as3_vsi4;

#define AS1(p) ((as1_vsi4*)(uintptr_t)(p))
#define AS3(p) ((as3_vsi4*)(uintptr_t)(p))   // low 32 bits of generic LDS addr == LDS offset

static __device__ __forceinline__ void wait_async_zero() {
#if HAVE_ASYNC_LDS
#if __has_builtin(__builtin_amdgcn_s_wait_asynccnt)
    __builtin_amdgcn_s_wait_asynccnt(0);
#else
    asm volatile("s_wait_asynccnt 0x0" ::: "memory");
#endif
#endif
}

static __device__ __forceinline__ unsigned pack2(float lo, float hi) {
#if __has_builtin(__builtin_amdgcn_cvt_pk_bf16_f32)
    v2bf t = __builtin_amdgcn_cvt_pk_bf16_f32(lo, hi);
    return __builtin_bit_cast(unsigned, t);
#else
    unsigned ul = __builtin_bit_cast(unsigned, lo);
    unsigned uh = __builtin_bit_cast(unsigned, hi);
    ul += 0x7FFFu + ((ul >> 16) & 1u);      // RNE to bf16
    uh += 0x7FFFu + ((uh >> 16) & 1u);
#if __has_builtin(__builtin_amdgcn_perm)
    return __builtin_amdgcn_perm(uh, ul, 0x07060302u);  // {uh[31:16], ul[31:16]}
#else
    return (ul >> 16) | (uh & 0xFFFF0000u);
#endif
#endif
}

// branch-free activations (v_exp_f32 + v_rcp_f32, saturate correctly at +/-inf)
static __device__ __forceinline__ float fast_sigmoid(float x) {
    return __builtin_amdgcn_rcpf(1.0f + __expf(-x));
}
static __device__ __forceinline__ float fast_tanh(float x) {
    return 1.0f - 2.0f * __builtin_amdgcn_rcpf(1.0f + __expf(2.0f * x));
}

// ---------------------------------------------------------------
// Kernel 1: convert the four fp32 weight matrices to bf16 in ws.
// Layout: wbf[g][n][k], g in {f,i,c,o}, n in [0,512), k in [0,1024)
// ---------------------------------------------------------------
__global__ void lstm_wconv(const float* __restrict__ Wf, const float* __restrict__ Wi,
                           const float* __restrict__ Wc, const float* __restrict__ Wo,
                           unsigned short* __restrict__ wbf) {
    const int e = (blockIdx.x * NTHREADS + threadIdx.x) * 8;   // 4*512*1024 / 8 threads
    const int gate = e >> 19;                                  // 2^19 elems per gate
    const int off  = e & ((1 << 19) - 1);
    const float* src = (gate == 0) ? Wf : (gate == 1) ? Wi : (gate == 2) ? Wc : Wo;
    float4 f0 = ((const float4*)(src + off))[0];
    float4 f1 = ((const float4*)(src + off))[1];
    uint4 u;
    u.x = pack2(f0.x, f0.y);
    u.y = pack2(f0.z, f0.w);
    u.z = pack2(f1.x, f1.y);
    u.w = pack2(f1.z, f1.w);
    *(uint4*)(wbf + e) = u;
}

// ---------------------------------------------------------------
// Kernel 2: fused 4-gate bf16 WMMA GEMM + LSTM epilogue.
// Double-buffered LDS; A register-prefetched (fp32->bf16 on the fly),
// B DMA'd global->LDS with async-to-LDS when available.
// grid = (HID/BN, BATCH/BM)
// ---------------------------------------------------------------
__global__ __launch_bounds__(NTHREADS, 1)
void lstm_wmma(const float* __restrict__ x, const float* __restrict__ h,
               const float* __restrict__ c,
               const float* __restrict__ bfp, const float* __restrict__ bip,
               const float* __restrict__ bcp, const float* __restrict__ bop,
               const unsigned short* __restrict__ wbf,
               float* __restrict__ out) {
    __shared__ unsigned short ldsA[2][BM * BK];          // 2 x  8 KB
    __shared__ unsigned short ldsB[2][4 * BN * BK];      // 2 x 16 KB

    const int tid  = threadIdx.x;
    const int lane = tid & 31;
    const int wave = tid >> 5;
    const int wm   = wave >> 1;        // 0..3  (M position, 32 rows each)
    const int wn   = wave & 1;         // 0..1  (N position, 32 cols each)
    const int l15  = lane & 15;
    const int lhi  = lane >> 4;        // which K half-group this lane holds

    const int rowBase = blockIdx.y * BM;   // batch offset
    const int colBase = blockIdx.x * BN;   // hidden-unit offset

    // cooperative-load indices
    const int ar = tid >> 1;               // A row      0..127
    const int ah = (tid & 1) * 16;         // A col half 0/16
    const int bg = tid >> 6;               // B gate     0..3
    const int bn = tid & 63;               // B row      0..63

    const unsigned short* wrow = wbf + ((size_t)bg << 19)
                                     + (size_t)(colBase + bn) * KTOT;
    unsigned short* ldsBrow0 = &ldsB[0][(bg * BN + bn) * BK];
    unsigned short* ldsBrow1 = &ldsB[1][(bg * BN + bn) * BK];

    v8f acc[4][2][2] = {};                 // 4 gates x (2x2) 16x16 f32 tiles

    float4 fa[4];                          // prefetched A chunk (16 fp32)

    // ---- prologue: A chunk0 -> regs, B chunk0 -> LDS[0] (async DMA if possible)
    {
        const float* srcA = x + (size_t)(rowBase + ar) * INF + ah;
        fa[0] = ((const float4*)srcA)[0];
        fa[1] = ((const float4*)srcA)[1];
        fa[2] = ((const float4*)srcA)[2];
        fa[3] = ((const float4*)srcA)[3];
#if HAVE_ASYNC_LDS
        __builtin_amdgcn_global_load_async_to_lds_b128(AS1(wrow),      AS3(ldsBrow0),      0, 0);
        __builtin_amdgcn_global_load_async_to_lds_b128(AS1(wrow + 8),  AS3(ldsBrow0 + 8),  0, 0);
        __builtin_amdgcn_global_load_async_to_lds_b128(AS1(wrow + 16), AS3(ldsBrow0 + 16), 0, 0);
        __builtin_amdgcn_global_load_async_to_lds_b128(AS1(wrow + 24), AS3(ldsBrow0 + 24), 0, 0);
#endif
    }

    int buf = 0;
    for (int k0 = 0; k0 < KTOT; k0 += BK) {
        unsigned short* ldsBrowCur = buf ? ldsBrow1 : ldsBrow0;
        unsigned short* ldsBrowNxt = buf ? ldsBrow0 : ldsBrow1;
        (void)ldsBrowCur;

        // ---- commit prefetched A chunk to LDS[buf] ----
        {
            uint4 u0, u1;
            u0.x = pack2(fa[0].x, fa[0].y); u0.y = pack2(fa[0].z, fa[0].w);
            u0.z = pack2(fa[1].x, fa[1].y); u0.w = pack2(fa[1].z, fa[1].w);
            u1.x = pack2(fa[2].x, fa[2].y); u1.y = pack2(fa[2].z, fa[2].w);
            u1.z = pack2(fa[3].x, fa[3].y); u1.w = pack2(fa[3].z, fa[3].w);
            unsigned short* dstA = &ldsA[buf][ar * BK + ah];
            ((uint4*)dstA)[0] = u0;
            ((uint4*)dstA)[1] = u1;
        }
#if !HAVE_ASYNC_LDS
        // fallback: synchronous B copy for the current chunk (transient regs)
        {
            const uint4* srcB = (const uint4*)(wrow + k0);
            uint4 w0 = srcB[0], w1 = srcB[1], w2 = srcB[2], w3 = srcB[3];
            ((uint4*)ldsBrowCur)[0] = w0;
            ((uint4*)ldsBrowCur)[1] = w1;
            ((uint4*)ldsBrowCur)[2] = w2;
            ((uint4*)ldsBrowCur)[3] = w3;
        }
#endif
        wait_async_zero();      // B[buf] DMA complete (no-op on fallback path)
        __syncthreads();

        // ---- issue next chunk's loads (hidden under the WMMAs) ----
        const int kn = k0 + BK;
        if (kn < KTOT) {
            const float* srcA = (kn < INF)
                ? (x + (size_t)(rowBase + ar) * INF + (kn + ah))
                : (h + (size_t)(rowBase + ar) * HID + (kn - INF + ah));
            fa[0] = ((const float4*)srcA)[0];
            fa[1] = ((const float4*)srcA)[1];
            fa[2] = ((const float4*)srcA)[2];
            fa[3] = ((const float4*)srcA)[3];
#if HAVE_ASYNC_LDS
            const unsigned short* srcB = wrow + kn;
            __builtin_amdgcn_global_load_async_to_lds_b128(AS1(srcB),      AS3(ldsBrowNxt),      0, 0);
            __builtin_amdgcn_global_load_async_to_lds_b128(AS1(srcB + 8),  AS3(ldsBrowNxt + 8),  0, 0);
            __builtin_amdgcn_global_load_async_to_lds_b128(AS1(srcB + 16), AS3(ldsBrowNxt + 16), 0, 0);
            __builtin_amdgcn_global_load_async_to_lds_b128(AS1(srcB + 24), AS3(ldsBrowNxt + 24), 0, 0);
#endif
        }

        // ---- A fragments (16x32 bf16 each): contiguous k = lhi*16 + s (as B), two b128 loads each
        AFrag a[2];
#pragma unroll
        for (int i = 0; i < 2; ++i) {
            const int row = wm * 32 + i * 16 + l15;
            const unsigned short* p = &ldsA[buf][row * BK + lhi * 16];
            a[i].h[0] = *(const v8bf*)(p);        // K = kb .. kb+7
            a[i].h[1] = *(const v8bf*)(p + 8);    // K = kb+8 .. kb+15
        }

        // ---- B fragments + 16 v_wmma_f32_16x16x32_bf16 per wave per chunk
#pragma unroll
        for (int g = 0; g < 4; ++g) {
#pragma unroll
            for (int j = 0; j < 2; ++j) {
                const int n = wn * 32 + j * 16 + l15;
                const unsigned short* p = &ldsB[buf][(g * BN + n) * BK + lhi * 16];
                v16bf b = *(const v16bf*)p;       // 16 contiguous halves
#pragma unroll
                for (int i = 0; i < 2; ++i) {
                    acc[g][i][j] = __builtin_amdgcn_wmma_f32_16x16x32_bf16(
                        false, a[i].v, false, b, (short)0, acc[g][i][j],
                        false, false);
                }
            }
        }
        buf ^= 1;
    }

    // ---- fused LSTM epilogue (branch-free activations) ----
    float bias[4][2];
#pragma unroll
    for (int j = 0; j < 2; ++j) {
        const int gc = colBase + wn * 32 + j * 16 + l15;
        bias[0][j] = bfp[gc];
        bias[1][j] = bip[gc];
        bias[2][j] = bcp[gc];
        bias[3][j] = bop[gc];
    }

    float* out_h = out;
    float* out_c = out + (size_t)BATCH * HID;

#pragma unroll
    for (int i = 0; i < 2; ++i) {
#pragma unroll
        for (int j = 0; j < 2; ++j) {
            const int gc = colBase + wn * 32 + j * 16 + l15;
#pragma unroll
            for (int r = 0; r < 8; ++r) {
                const int gr = rowBase + wm * 32 + i * 16 + lhi * 8 + r;
                const float vf = fast_sigmoid(acc[0][i][j][r] + bias[0][j]);
                const float vi = fast_sigmoid(acc[1][i][j][r] + bias[1][j]);
                const float vg = fast_tanh  (acc[2][i][j][r] + bias[2][j]);
                const float vo = fast_sigmoid(acc[3][i][j][r] + bias[3][j]);
                const float cold = c[(size_t)gr * HID + gc];
                const float nc = fast_tanh(vf * cold + vi * vg);
                const float nh = nc * vo;
                out_h[(size_t)gr * HID + gc] = nh;
                out_c[(size_t)gr * HID + gc] = nc;
            }
        }
    }
}

extern "C" void kernel_launch(void* const* d_in, const int* in_sizes, int n_in,
                              void* d_out, int out_size, void* d_ws, size_t ws_size,
                              hipStream_t stream) {
    (void)in_sizes; (void)n_in; (void)out_size; (void)ws_size;
    const float* x  = (const float*)d_in[0];
    const float* h  = (const float*)d_in[1];
    const float* c  = (const float*)d_in[2];
    const float* Wf = (const float*)d_in[3];
    const float* bf = (const float*)d_in[4];
    const float* Wi = (const float*)d_in[5];
    const float* bi = (const float*)d_in[6];
    const float* Wc = (const float*)d_in[7];
    const float* bc = (const float*)d_in[8];
    const float* Wo = (const float*)d_in[9];
    const float* bo = (const float*)d_in[10];

    unsigned short* wbf = (unsigned short*)d_ws;   // 4 * 512 * 1024 bf16 = 4 MB

    // weights -> bf16 (once per launch; deterministic)
    const int convThreads = 4 * HID * KTOT / 8;    // 262144
    lstm_wconv<<<convThreads / NTHREADS, NTHREADS, 0, stream>>>(Wf, Wi, Wc, Wo, wbf);

    dim3 grid(HID / BN, BATCH / BM);               // (8, 128)
    lstm_wmma<<<grid, NTHREADS, 0, stream>>>(x, h, c, bf, bi, bc, bo, wbf,
                                             (float*)d_out);
}
